// EncoderDecoderVAE_62268435857718
// MI455X (gfx1250) — matrix operand order repaired and checked by reading, not compiled, not run on hardware
//
#include <hip/hip_runtime.h>
#include <hip/hip_bf16.h>

// ---------------------------------------------------------------------------
// EncoderDecoderVAE for MI455X (gfx1250, wave32, WMMA f32<-f16 16x16x32,
// async global->LDS staging in the MMD Gram kernel).
//
// d_out layout (fp32): recon[8192*1024] | latent[8192*64] | reg_loss[1]
// d_ws layout (bytes):
//   0        WencT  f16 [64][1024]   (131072 B)
//   131072   WdecT  f16 [1024][64]   (131072 B)
//   262144   x16    f16 [8192][64]   (1048576 B)
//   1310720  y16    f16 [8192][64]   (1048576 B)
//   2359296  nx     f32 [8192]       (32768 B)
//   2392064  ny     f32 [8192]       (32768 B)
//   2424832  partials f32 [3][4096]  (49152 B)   total ~2.36 MB
// ---------------------------------------------------------------------------

typedef _Float16 h16_t;
typedef __attribute__((ext_vector_type(16))) _Float16 v16h;
typedef __attribute__((ext_vector_type(8)))  _Float16 v8h;
typedef __attribute__((ext_vector_type(8)))  float    v8f;
typedef __attribute__((ext_vector_type(4)))  float    v4f;

#define NROWS   8192
#define DMODEL  1024
#define DLAT    64
#define YLDS_STRIDE 72   // halves; 144 B row stride -> conflict-free ds_load_b128

static __device__ __forceinline__ v8f wmma_f16(v16h a, v16h b, v8f c) {
  return __builtin_amdgcn_wmma_f32_16x16x32_f16(
      /*neg_a=*/false, a, /*neg_b=*/false, b,
      /*c_mod=*/(short)0, c, /*reuse_a=*/false, /*reuse_b=*/false);
}

// 32-bit LDS byte offset of a generic pointer known to be in LDS.
static __device__ __forceinline__ unsigned lds_addr_of(const void* p) {
  return (unsigned)(uintptr_t)(__attribute__((address_space(3))) const void*)p;
}

// A fragment (16x32 f16, M x K) from a row-major fp32 matrix, converting on
// the fly.  ISA layout: lanes 0-15 hold M=lane; halves 0-7 -> K=k0+hi*8..+7,
// halves 8-15 -> K=k0+16+hi*8..+7 (hi = lane>>4).
static __device__ __forceinline__ v16h
load_a_frag_f32(const float* __restrict__ base, int ld, int r0, int k0, int lane) {
  const int l = lane & 15, hi = lane >> 4;
  const float* p = base + (size_t)(r0 + l) * ld + k0 + hi * 8;
  v4f f0 = *(const v4f*)(p);
  v4f f1 = *(const v4f*)(p + 4);
  v4f f2 = *(const v4f*)(p + 16);
  v4f f3 = *(const v4f*)(p + 20);
  v16h a;
#pragma unroll
  for (int i = 0; i < 4; ++i) {
    a[i]      = (_Float16)f0[i];
    a[4 + i]  = (_Float16)f1[i];
    a[8 + i]  = (_Float16)f2[i];
    a[12 + i] = (_Float16)f3[i];
  }
  return a;
}

// A fragment from a row-major f16 matrix (two 16B contiguous loads per lane).
static __device__ __forceinline__ v16h
load_a_frag_f16(const h16_t* __restrict__ base, int ld, int r0, int k0, int lane) {
  const int l = lane & 15, hi = lane >> 4;
  const h16_t* p = base + (size_t)(r0 + l) * ld + k0 + hi * 8;
  union { v16h v; v8h h[2]; } u;
  u.h[0] = *(const v8h*)(p);
  u.h[1] = *(const v8h*)(p + 16);
  return u.v;
}

// B fragment (32x16 f16, K x N) from a *transposed* f16 matrix WT[N][K]:
// lane holds column c0+(lane&15); halves 0-7 -> K = k0 + hi*8 + j, halves 8-15
// -> K = k0 + 16 + hi*8 + (j-8), contiguous runs in WT's row -> two 16B loads.
static __device__ __forceinline__ v16h
load_b_frag_T(const h16_t* __restrict__ baseT, int ldk, int c0, int k0, int lane) {
  const int l = lane & 15, hi = lane >> 4;
  const h16_t* p = baseT + (size_t)(c0 + l) * ldk + k0 + hi * 8;
  union { v16h v; v8h h[2]; } u;
  u.h[0] = *(const v8h*)(p);
  u.h[1] = *(const v8h*)(p + 16);
  return u.v;
}

// B fragment from the padded LDS y-tile (ds_load_b128 x2, conflict-free).
static __device__ __forceinline__ v16h
load_b_frag_lds(const h16_t* sm, int c0, int k0, int lane) {
  const int l = lane & 15, hi = lane >> 4;
  const h16_t* p = sm + (c0 + l) * YLDS_STRIDE + k0 + hi * 8;
  union { v16h v; v8h h[2]; } u;
  u.h[0] = *(const v8h*)(p);
  u.h[1] = *(const v8h*)(p + 16);
  return u.v;
}

// ---------------------------------------------------------------------------
// Prep: convert + transpose a row-major fp32 [K][N] into f16 [N][K].
__global__ void k_cvt_transpose(const float* __restrict__ W, h16_t* __restrict__ WT,
                                int K, int N) {
  int idx = blockIdx.x * blockDim.x + threadIdx.x;
  const int total = K * N;
  for (; idx < total; idx += gridDim.x * blockDim.x) {
    const int k = idx / N, n = idx - k * N;
    WT[(size_t)n * K + k] = (h16_t)W[idx];
  }
}

// One wave per row of a [8192][64] fp32 matrix: squared-norm (+ optional f16
// conversion).  Lane loads 2 contiguous floats; shfl tree-reduce (wave32).
__global__ void __launch_bounds__(256)
k_row_norm(const float* __restrict__ src, float* __restrict__ norms,
           h16_t* __restrict__ dst16, int do_cvt) {
  const int wave = threadIdx.x >> 5, lane = threadIdx.x & 31;
  const int row = blockIdx.x * 8 + wave;
  const float* p = src + (size_t)row * DLAT + lane * 2;
  const float a = p[0], b = p[1];
  if (do_cvt) {
    dst16[(size_t)row * DLAT + lane * 2]     = (h16_t)a;
    dst16[(size_t)row * DLAT + lane * 2 + 1] = (h16_t)b;
  }
  float s = a * a + b * b;
#pragma unroll
  for (int off = 16; off; off >>= 1) s += __shfl_down(s, off);
  if (lane == 0) norms[row] = s;
}

// ---------------------------------------------------------------------------
// Encoder: latent = X[8192][1024] @ W_enc[1024][64] + b_enc.
// Block = 8 waves; wave w owns rows [bx*128 + 16w, +16), all 4 col tiles.
// A fragment is software-pipelined one k-step ahead; B loads batched before
// the WMMA group so the clause latency overlaps the matrix pipe.
__global__ void __launch_bounds__(256)
k_encoder(const float* __restrict__ X, const h16_t* __restrict__ WencT,
          const float* __restrict__ b_enc,
          float* __restrict__ latent, h16_t* __restrict__ y16) {
  const int wave = threadIdx.x >> 5, lane = threadIdx.x & 31;
  const int r0 = blockIdx.x * 128 + wave * 16;
  const int l = lane & 15, hi = lane >> 4;

  v8f acc[4] = {};
  v16h a_cur = load_a_frag_f32(X, DMODEL, r0, 0, lane);
  for (int k0 = 0; k0 < DMODEL; k0 += 32) {
    const v16h b0 = load_b_frag_T(WencT, DMODEL, 0,  k0, lane);
    const v16h b1 = load_b_frag_T(WencT, DMODEL, 16, k0, lane);
    const v16h b2 = load_b_frag_T(WencT, DMODEL, 32, k0, lane);
    const v16h b3 = load_b_frag_T(WencT, DMODEL, 48, k0, lane);
    v16h a_nxt = a_cur;
    if (k0 + 32 < DMODEL) {
      __builtin_prefetch(X + (size_t)(r0 + l) * DMODEL + k0 + 64, 0, 1);
      a_nxt = load_a_frag_f32(X, DMODEL, r0, k0 + 32, lane);
    }
    acc[0] = wmma_f16(a_cur, b0, acc[0]);
    acc[1] = wmma_f16(a_cur, b1, acc[1]);
    acc[2] = wmma_f16(a_cur, b2, acc[2]);
    acc[3] = wmma_f16(a_cur, b3, acc[3]);
    a_cur = a_nxt;
  }
#pragma unroll
  for (int t = 0; t < 4; ++t) {
    const int c = t * 16 + l;
    const float bias = b_enc[c];
#pragma unroll
    for (int v = 0; v < 8; ++v) {
      const int row = r0 + v + 8 * hi;   // C layout: VGPR v -> M = v + 8*hi
      const float val = acc[t][v] + bias;
      latent[(size_t)row * DLAT + c] = val;
      y16[(size_t)row * DLAT + c]    = (h16_t)val;
    }
  }
}

// ---------------------------------------------------------------------------
// Decoder: recon = latent16[8192][64] @ W_dec[64][1024] + b_dec.
// Block covers 128 rows x 256 cols; wave w: one 16-row slab, 16 col tiles,
// K=64 -> 2 WMMA per tile (A frags loaded once, reused 16x).
__global__ void __launch_bounds__(256)
k_decoder(const h16_t* __restrict__ y16, const h16_t* __restrict__ WdecT,
          const float* __restrict__ b_dec, float* __restrict__ recon) {
  const int wave = threadIdx.x >> 5, lane = threadIdx.x & 31;
  const int r0 = blockIdx.x * 128 + wave * 16;
  const int cbase = blockIdx.y * 256;
  const int l = lane & 15, hi = lane >> 4;

  const v16h a0 = load_a_frag_f16(y16, DLAT, r0, 0, lane);
  const v16h a1 = load_a_frag_f16(y16, DLAT, r0, 32, lane);

#pragma unroll 4
  for (int t = 0; t < 16; ++t) {
    const int c0 = cbase + t * 16;
    const v16h b0 = load_b_frag_T(WdecT, DLAT, c0, 0, lane);
    const v16h b1 = load_b_frag_T(WdecT, DLAT, c0, 32, lane);
    v8f acc = {};
    acc = wmma_f16(a0, b0, acc);
    acc = wmma_f16(a1, b1, acc);
    const float bias = b_dec[c0 + l];
#pragma unroll
    for (int v = 0; v < 8; ++v)
      recon[(size_t)(r0 + v + 8 * hi) * DMODEL + c0 + l] = acc[v] + bias;
  }
}

// ---------------------------------------------------------------------------
// MMD tile: block computes a 128x128 tile of exp(-(nx_i + ny_j - 2 x_i.y_j)/4096)
// and emits one partial sum per block (deterministic; no atomics).
//
// The 128x64 f16 y-tile (16 KB) is shared by all 8 waves, so it is staged into
// LDS once per block via GLOBAL_LOAD_ASYNC_TO_LDS_B128 (ASYNCcnt path), with a
// padded row stride (144 B) so the ds_load_b128 B-fragment reads start on 16
// distinct 4-bank groups (conflict-free).
__global__ void __launch_bounds__(256)
k_mmd_tile(const h16_t* __restrict__ xf, const h16_t* __restrict__ yf,
           const float* __restrict__ nx, const float* __restrict__ ny,
           float* __restrict__ partial) {
  __shared__ h16_t ytile[128 * YLDS_STRIDE];   // 18 KB
  __shared__ float wsum[8];
  const int tid = threadIdx.x;
  const int wave = tid >> 5, lane = tid & 31;
  const int l = lane & 15, hi = lane >> 4;
  const int r0 = blockIdx.x * 128 + wave * 16;
  const int cbase = blockIdx.y * 128;

  // ---- async-stage y tile: 1024 chunks of 16 B, 4 per thread -------------
  {
    const unsigned lds_base = lds_addr_of(&ytile[0]);
#pragma unroll
    for (int i = 0; i < 4; ++i) {
      const int chunk = tid + i * 256;              // 0..1023
      const int row = chunk >> 3, seg = chunk & 7;  // row<128, seg<8
      const h16_t* g = yf + (size_t)(cbase + row) * DLAT + seg * 8;
      const unsigned ldsa = lds_base + (unsigned)(row * (YLDS_STRIDE * 2) + seg * 16);
      asm volatile("global_load_async_to_lds_b128 %0, %1, off"
                   :: "v"(ldsa), "v"((unsigned long long)g)
                   : "memory");
    }
    asm volatile("s_wait_asynccnt 0" ::: "memory");
  }

  // A fragments + row norms (private to the wave) while the barrier settles.
  const v16h a0 = load_a_frag_f16(xf, DLAT, r0, 0, lane);
  const v16h a1 = load_a_frag_f16(xf, DLAT, r0, 32, lane);
  float rn[8];
#pragma unroll
  for (int v = 0; v < 8; ++v) rn[v] = nx[r0 + v + 8 * hi];

  __syncthreads();   // all waves' async copies complete -> y tile visible

  const float scale = -1.0f / 4096.0f;   // dim*dim = 64*64
  float s = 0.0f;
#pragma unroll 2
  for (int t = 0; t < 8; ++t) {
    const int c0 = t * 16;               // column tile, local to the y tile
    const v16h b0 = load_b_frag_lds(ytile, c0, 0, lane);
    const v16h b1 = load_b_frag_lds(ytile, c0, 32, lane);
    v8f acc = {};
    acc = wmma_f16(a0, b0, acc);
    acc = wmma_f16(a1, b1, acc);
    const float cn = ny[cbase + c0 + l];
#pragma unroll
    for (int v = 0; v < 8; ++v) {
      const float sq = rn[v] + cn - 2.0f * acc[v];
      s += __expf(sq * scale);           // v_exp_f32
    }
  }
#pragma unroll
  for (int off = 16; off; off >>= 1) s += __shfl_down(s, off);
  if (lane == 0) wsum[wave] = s;
  __syncthreads();
  if (tid == 0) {
    float t = 0.0f;
#pragma unroll
    for (int i = 0; i < 8; ++i) t += wsum[i];
    partial[blockIdx.y * gridDim.x + blockIdx.x] = t;
  }
}

// Fixed-order reduction of the 3 partial arrays -> scalar loss (deterministic).
__global__ void __launch_bounds__(128)
k_reduce_finalize(const float* __restrict__ partials, float* __restrict__ out) {
  __shared__ float res[3];
  const int wave = threadIdx.x >> 5, lane = threadIdx.x & 31;
  if (wave < 3) {
    const float* p = partials + wave * 4096;
    float s = 0.0f;
    for (int i = lane; i < 4096; i += 32) s += p[i];
#pragma unroll
    for (int off = 16; off; off >>= 1) s += __shfl_down(s, off);
    if (lane == 0) res[wave] = s;
  }
  __syncthreads();
  if (threadIdx.x == 0) {
    const float m = 1.0f / ((float)NROWS * (float)NROWS);
    out[0] = (res[0] + res[1] - 2.0f * res[2]) * m;
  }
}

// ---------------------------------------------------------------------------
extern "C" void kernel_launch(void* const* d_in, const int* in_sizes, int n_in,
                              void* d_out, int out_size, void* d_ws, size_t ws_size,
                              hipStream_t stream) {
  (void)in_sizes; (void)n_in; (void)out_size; (void)ws_size;

  const float* X  = (const float*)d_in[0];  // input_encoding [8192][1024]
  const float* TS = (const float*)d_in[1];  // true_samples   [8192][64]
  const float* We = (const float*)d_in[2];  // W_enc          [1024][64]
  const float* be = (const float*)d_in[3];  // b_enc          [64]
  const float* Wd = (const float*)d_in[4];  // W_dec          [64][1024]
  const float* bd = (const float*)d_in[5];  // b_dec          [1024]

  float* recon  = (float*)d_out;                         // [8192][1024]
  float* latent = recon + (size_t)NROWS * DMODEL;        // [8192][64]
  float* loss   = latent + (size_t)NROWS * DLAT;         // [1]

  char*  ws      = (char*)d_ws;
  h16_t* WencT   = (h16_t*)(ws);                         // [64][1024]
  h16_t* WdecT   = (h16_t*)(ws + 131072);                // [1024][64]
  h16_t* x16     = (h16_t*)(ws + 262144);                // [8192][64]
  h16_t* y16     = (h16_t*)(ws + 262144 + 1048576);      // [8192][64]
  float* nx      = (float*)(ws + 262144 + 2 * 1048576);  // [8192]
  float* ny      = nx + NROWS;                           // [8192]
  float* parts   = ny + NROWS;                           // [3][4096]

  k_cvt_transpose<<<256, 256, 0, stream>>>(We, WencT, DMODEL, DLAT);
  k_cvt_transpose<<<256, 256, 0, stream>>>(Wd, WdecT, DLAT, DMODEL);
  k_row_norm<<<NROWS / 8, 256, 0, stream>>>(TS, nx, x16, 1);

  k_encoder<<<NROWS / 128, 256, 0, stream>>>(X, WencT, be, latent, y16);
  k_row_norm<<<NROWS / 8, 256, 0, stream>>>(latent, ny, y16, 0);
  k_decoder<<<dim3(NROWS / 128, DMODEL / 256), 256, 0, stream>>>(y16, WdecT, bd, recon);

  k_mmd_tile<<<dim3(64, 64), 256, 0, stream>>>(x16, x16, nx, nx, parts);
  k_mmd_tile<<<dim3(64, 64), 256, 0, stream>>>(y16, y16, ny, ny, parts + 4096);
  k_mmd_tile<<<dim3(64, 64), 256, 0, stream>>>(x16, y16, nx, ny, parts + 8192);

  k_reduce_finalize<<<1, 128, 0, stream>>>(parts, loss);
}
